// inference_model_10969346474917
// MI455X (gfx1250) — compile-verified, matrix operand order checked by reading, not run on hardware
//
#include <hip/hip_runtime.h>
#include <hip/hip_bf16.h>

#define B_   32
#define EVI  5
#define L_   130
#define LP   144        // L padded to 9*16
#define H_   768
#define K_   21
#define NL_  3
#define N_   (B_*EVI)
#define QT   9          // ceil(130/16)

typedef __attribute__((ext_vector_type(16))) _Float16 v16h;
typedef __attribute__((ext_vector_type(8)))  _Float16 v8h;
typedef __attribute__((ext_vector_type(8)))  float    v8f;

// kernel mus: [1.0, 0.95, 0.85, ..., -0.95]; inv2s2 = 0.5/sigma^2
__constant__ float c_mu[K_] = {
    1.0f, 0.95f, 0.85f, 0.75f, 0.65f, 0.55f, 0.45f, 0.35f, 0.25f, 0.15f, 0.05f,
    -0.05f,-0.15f,-0.25f,-0.35f,-0.45f,-0.55f,-0.65f,-0.75f,-0.85f,-0.95f};
__constant__ float c_i2s2[K_] = {
    5.0e5f, 50.f,50.f,50.f,50.f,50.f,50.f,50.f,50.f,50.f,50.f,
    50.f,50.f,50.f,50.f,50.f,50.f,50.f,50.f,50.f,50.f};

// ---------------- init: zero the atomic accumulator ----------------
__global__ void k_init(float* selnum) {
    int t = threadIdx.x + blockIdx.x * blockDim.x;
    if (t < N_) selnum[t] = 0.f;
}

// ---------------- masks: m_text / m_ev / m_claim (padded to LP) + claim sums
__global__ __launch_bounds__(256)
void k_masks(const int* __restrict__ msk, const int* __restrict__ seg,
             float* __restrict__ m_text, float* __restrict__ m_ev,
             float* __restrict__ m_claim, float* __restrict__ claimsum) {
    const int n = blockIdx.x;
    const int t = threadIdx.x;
    __shared__ float red[256];
    float mc = 0.f;
    if (t < LP) {
        float mt = 0.f, me = 0.f;
        if (t > 0 && t < L_) {
            mt = (float)msk[n * L_ + t];
            float sg = (float)seg[n * L_ + t];
            me = sg * mt;
            mc = (1.f - sg) * mt;
        }
        m_text[n * LP + t] = mt;
        m_ev[n * LP + t]   = me;
        m_claim[n * LP + t] = mc;
    }
    red[t] = mc; __syncthreads();
    for (int s = 128; s > 0; s >>= 1) { if (t < s) red[t] += red[t + s]; __syncthreads(); }
    if (t == 0) claimsum[n] = red[0];
}

// ---------------- l2 normalize each (n,l) row and convert to f16 ----------
__global__ __launch_bounds__(256)
void k_norm(const float* __restrict__ hiddens, _Float16* __restrict__ hn) {
    const size_t row = blockIdx.x;
    const int t = threadIdx.x;
    const float* x = hiddens + row * H_;
    float ss = 0.f;
    for (int h = t; h < H_; h += 256) { float v = x[h]; ss += v * v; }
    __shared__ float red[256];
    red[t] = ss; __syncthreads();
    for (int s = 128; s > 0; s >>= 1) { if (t < s) red[t] += red[t + s]; __syncthreads(); }
    const float inv = 1.f / fmaxf(sqrtf(red[0]), 1e-12f);
    for (int h = t; h < H_; h += 256) hn[row * H_ + h] = (_Float16)(x[h] * inv);
}

// ---------------- WMMA pairwise sim + fused Gaussian kernel pooling -------
// one wave per (pair=(b,e,i), 16-query tile). A = key rows (evidence i),
// B = query rows (evidence e). D[m=key, n=query]. Pool over keys, emit
// fused attention score and (diag pairs) the sel numerator.
__global__ __launch_bounds__(32)
void k_pairpool(const _Float16* __restrict__ hn,
                const float* __restrict__ m_text,
                const float* __restrict__ m_ev,
                const float* __restrict__ m_claim,
                const float* __restrict__ W_att, const float* __restrict__ b_att,
                const float* __restrict__ W_sel,
                float* __restrict__ score, float* __restrict__ selnum) {
    const int qt = blockIdx.x;
    const int e  = blockIdx.y / EVI;
    const int i  = blockIdx.y % EVI;
    const int b  = blockIdx.z;
    const int ne = b * EVI + e;
    const int ni = b * EVI + i;
    const int pair = (b * EVI + e) * EVI + i;
    const bool diag = (e == i);
    const int lane = threadIdx.x;
    const int hlf  = lane >> 4;           // which lane half
    const int mrow = lane & 15;
    const int qbase = qt * 16;

    __shared__ float smt[LP];   // key-side text mask (evidence i)
    __shared__ float sme[LP];   // key-side evidence mask
    for (int l = lane; l < LP; l += 32) {
        smt[l] = m_text[ni * LP + l];
        sme[l] = m_ev[ni * LP + l];
    }
    __syncthreads();

    // B-fragment pointer (query rows of e), fixed for this block
    int qr = qbase + mrow; if (qr > L_ - 1) qr = L_ - 1;
    const _Float16* bp = hn + ((size_t)(ne * L_ + qr)) * H_ + hlf * 8;

    float pool[K_], psel[K_];
    #pragma unroll
    for (int k = 0; k < K_; ++k) { pool[k] = 0.f; psel[k] = 0.f; }

    for (int dt = 0; dt < QT; ++dt) {
        const int dbase = dt * 16;
        int dr = dbase + mrow; if (dr > L_ - 1) dr = L_ - 1;
        const _Float16* ap = hn + ((size_t)(ni * L_ + dr)) * H_ + hlf * 8;
        // speculative prefetch of next key tile -> global_prefetch_b8
        __builtin_prefetch((const void*)(ap + 16 * (size_t)H_), 0, 0);

        v8f c = {};
        for (int h0 = 0; h0 < H_; h0 += 32) {
            union { v16h v; v8h h[2]; } a, bb;
            a.h[0]  = *(const v8h*)(ap + h0);
            a.h[1]  = *(const v8h*)(ap + h0 + 16);
            bb.h[0] = *(const v8h*)(bp + h0);
            bb.h[1] = *(const v8h*)(bp + h0 + 16);
            c = __builtin_amdgcn_wmma_f32_16x16x32_f16(
                    false, a.v, false, bb.v, (short)0, c, false, false);
        }
        // Gaussian pooling over this key tile
        #pragma unroll
        for (int v = 0; v < 8; ++v) {
            const int key = dbase + v + (hlf ? 8 : 0);
            const float s  = c[v];
            const float wt = smt[key];
            const float we = diag ? sme[key] : 0.f;
            #pragma unroll
            for (int k = 0; k < K_; ++k) {
                float d = s - c_mu[k];
                float g = __expf(-d * d * c_i2s2[k]);
                pool[k] += wt * g;
                psel[k] += we * g;
            }
        }
    }

    // combine the two lane halves (key rows m and m+8 live in lane^16)
    #pragma unroll
    for (int k = 0; k < K_; ++k) pool[k] += __shfl_xor(pool[k], 16, 32);
    float sc = b_att[0];
    #pragma unroll
    for (int k = 0; k < K_; ++k) sc += W_att[k] * __logf(fmaxf(pool[k], 1e-10f));

    const int q = qbase + mrow;
    if (diag) {
        #pragma unroll
        for (int k = 0; k < K_; ++k) psel[k] += __shfl_xor(psel[k], 16, 32);
        float sq = 0.f;
        #pragma unroll
        for (int k = 0; k < K_; ++k) sq += W_sel[k] * __logf(fmaxf(psel[k], 1e-10f));
        if (hlf == 0 && q < L_) {
            float mc = m_claim[ne * LP + q];
            if (mc > 0.f) atomicAdd(&selnum[ne], sq * mc);
        }
    }
    if (hlf == 0 && q < L_) {
        float mt = m_text[ne * LP + q];
        score[pair * L_ + q] = (mt > 0.f) ? sc : -10000.0f;
    }
}

// ---------------- masked softmax over q + denoise GEMV --------------------
__global__ __launch_bounds__(256)
void k_attn_denoise(const float* __restrict__ score,
                    const float* __restrict__ hiddens,
                    float* __restrict__ denoise) {
    const int p = blockIdx.x;            // flat (b,i,e)
    const int e = p % EVI;
    const int i = (p / EVI) % EVI;
    const int b = p / (EVI * EVI);
    const int pair = (b * EVI + e) * EVI + i;
    const int ne = b * EVI + e;
    const int t = threadIdx.x;
    __shared__ float att[L_];
    __shared__ float red[256];
    float sc = (t < L_) ? score[pair * L_ + t] : -3.0e38f;
    red[t] = sc; __syncthreads();
    for (int s = 128; s > 0; s >>= 1) { if (t < s) red[t] = fmaxf(red[t], red[t + s]); __syncthreads(); }
    const float mx = red[0]; __syncthreads();
    float ex = (t < L_) ? __expf(sc - mx) : 0.f;
    red[t] = ex; __syncthreads();
    for (int s = 128; s > 0; s >>= 1) { if (t < s) red[t] += red[t + s]; __syncthreads(); }
    const float inv = 1.f / red[0];
    if (t < L_) att[t] = ex * inv;
    __syncthreads();
    const float* hb = hiddens + (size_t)ne * L_ * H_;
    for (int h = t; h < H_; h += 256) {
        float acc = 0.f;
        for (int q = 0; q < L_; ++q) acc += att[q] * hb[q * H_ + h];
        denoise[(size_t)p * H_ + h] = acc;
    }
}

// ---------------- gating MLP: relu(Wg1 x + bg1) . Wg2 + bg2 ---------------
__global__ __launch_bounds__(128)
void k_gate(const float* __restrict__ pooled, const float* __restrict__ denoise,
            const float* __restrict__ Wg1, const float* __restrict__ bg1,
            const float* __restrict__ Wg2, const float* __restrict__ bg2,
            float* __restrict__ gate) {
    const int p = blockIdx.x;            // flat (b,i,e)
    const int i = (p / EVI) % EVI;
    const int b = p / (EVI * EVI);
    const int t = threadIdx.x;           // one hidden unit per thread
    const float* x1 = pooled + (size_t)(b * EVI + i) * H_;
    const float* x2 = denoise + (size_t)p * H_;
    const float* wr = Wg1 + (size_t)t * (2 * H_);
    float acc = bg1[t];
    for (int h = 0; h < H_; ++h) acc += wr[h] * x1[h];
    for (int h = 0; h < H_; ++h) acc += wr[H_ + h] * x2[h];
    float g = fmaxf(acc, 0.f) * Wg2[t];
    __shared__ float red[128];
    red[t] = g; __syncthreads();
    for (int s = 64; s > 0; s >>= 1) { if (t < s) red[t] += red[t + s]; __syncthreads(); }
    if (t == 0) gate[p] = red[0] + bg2[0];
}

// ---------------- sel logits + softmax over evidence ----------------------
__global__ __launch_bounds__(256)
void k_selfinal(const float* __restrict__ selnum, const float* __restrict__ claimsum,
                const float* __restrict__ b_sel, float* __restrict__ selprob) {
    const int t = threadIdx.x;
    __shared__ float logit[N_];
    if (t < N_) logit[t] = selnum[t] / (claimsum[t] + 1e-10f) + b_sel[0];
    __syncthreads();
    if (t < B_) {
        float mx = -3.0e38f;
        for (int i = 0; i < EVI; ++i) mx = fmaxf(mx, logit[t * EVI + i]);
        float ex[EVI]; float s = 0.f;
        for (int i = 0; i < EVI; ++i) { ex[i] = __expf(logit[t * EVI + i] - mx); s += ex[i]; }
        for (int i = 0; i < EVI; ++i) selprob[t * EVI + i] = ex[i] / s;
    }
}

// ---------------- gate softmax over e + weighted denoise sum --------------
__global__ __launch_bounds__(256)
void k_decombine(const float* __restrict__ gate, const float* __restrict__ denoise,
                 float* __restrict__ de) {
    const int bi = blockIdx.x;           // (b,i)
    const int t = threadIdx.x;
    float g[EVI], w[EVI];
    float mx = -3.0e38f;
    for (int e2 = 0; e2 < EVI; ++e2) { g[e2] = gate[bi * EVI + e2]; mx = fmaxf(mx, g[e2]); }
    float s = 0.f;
    for (int e2 = 0; e2 < EVI; ++e2) { w[e2] = __expf(g[e2] - mx); s += w[e2]; }
    const float inv = 1.f / s;
    for (int h = t; h < H_; h += 256) {
        float acc = 0.f;
        for (int e2 = 0; e2 < EVI; ++e2)
            acc += w[e2] * inv * denoise[((size_t)bi * EVI + e2) * H_ + h];
        de[(size_t)bi * H_ + h] = acc;
    }
}

// ---------------- final linear head: feat = Wd [pb; de] + bd --------------
__global__ __launch_bounds__(96)
void k_feat(const float* __restrict__ pooled, const float* __restrict__ de,
            const float* __restrict__ Wd, const float* __restrict__ bd,
            float* __restrict__ feat) {
    const int bi = blockIdx.x;           // (b,i)
    const int c = threadIdx.x >> 5;      // class row, one wave each
    const int lane = threadIdx.x & 31;
    const float* w = Wd + (size_t)c * (2 * H_);
    float acc = 0.f;
    for (int h = lane; h < H_; h += 32) acc += w[h] * pooled[(size_t)bi * H_ + h];
    for (int h = lane; h < H_; h += 32) acc += w[H_ + h] * de[(size_t)bi * H_ + h];
    for (int off = 16; off > 0; off >>= 1) acc += __shfl_down(acc, off, 32);
    if (lane == 0) feat[bi * NL_ + c] = acc + bd[c];
}

// ---------------- mixture of class softmaxes -> log prob ------------------
__global__ __launch_bounds__(32)
void k_final(const float* __restrict__ feat, const float* __restrict__ selprob,
             float* __restrict__ out) {
    const int b = blockIdx.x;
    if (threadIdx.x != 0) return;
    float pr[NL_] = {0.f, 0.f, 0.f};
    for (int i = 0; i < EVI; ++i) {
        const float* f = feat + (size_t)(b * EVI + i) * NL_;
        float mx = fmaxf(f[0], fmaxf(f[1], f[2]));
        float e0 = __expf(f[0] - mx), e1 = __expf(f[1] - mx), e2 = __expf(f[2] - mx);
        float inv = 1.f / (e0 + e1 + e2);
        float sp = selprob[b * EVI + i];
        pr[0] += sp * e0 * inv;
        pr[1] += sp * e1 * inv;
        pr[2] += sp * e2 * inv;
    }
    out[b * NL_ + 0] = __logf(pr[0]);
    out[b * NL_ + 1] = __logf(pr[1]);
    out[b * NL_ + 2] = __logf(pr[2]);
}

extern "C" void kernel_launch(void* const* d_in, const int* in_sizes, int n_in,
                              void* d_out, int out_size, void* d_ws, size_t ws_size,
                              hipStream_t stream) {
    (void)in_sizes; (void)n_in; (void)out_size; (void)ws_size;
    const float* hiddens = (const float*)d_in[0];
    const float* pooled  = (const float*)d_in[1];
    const int*   msk     = (const int*)d_in[2];
    const int*   seg     = (const int*)d_in[3];
    const float* W_att   = (const float*)d_in[4];
    const float* b_att   = (const float*)d_in[5];
    const float* W_sel   = (const float*)d_in[6];
    const float* b_sel   = (const float*)d_in[7];
    const float* Wg1     = (const float*)d_in[8];
    const float* bg1     = (const float*)d_in[9];
    const float* Wg2     = (const float*)d_in[10];
    const float* bg2     = (const float*)d_in[11];
    const float* Wd      = (const float*)d_in[12];
    const float* bd      = (const float*)d_in[13];
    float* out = (float*)d_out;

    char* ws = (char*)d_ws;
    size_t off = 0;
    auto take = [&](size_t bytes) -> char* {
        char* p = ws + off;
        off = (off + bytes + 255) & ~(size_t)255;
        return p;
    };
    _Float16* hn    = (_Float16*)take((size_t)N_ * L_ * H_ * 2);   // ~32 MB
    float* m_text   = (float*)take((size_t)N_ * LP * 4);
    float* m_ev     = (float*)take((size_t)N_ * LP * 4);
    float* m_claim  = (float*)take((size_t)N_ * LP * 4);
    float* claimsum = (float*)take((size_t)N_ * 4);
    float* score    = (float*)take((size_t)B_ * EVI * EVI * L_ * 4);
    float* selnum   = (float*)take((size_t)N_ * 4);
    float* selprob  = (float*)take((size_t)N_ * 4);
    float* denoise  = (float*)take((size_t)B_ * EVI * EVI * H_ * 4);
    float* gate     = (float*)take((size_t)B_ * EVI * EVI * 4);
    float* de       = (float*)take((size_t)N_ * H_ * 4);
    float* feat     = (float*)take((size_t)N_ * NL_ * 4);

    k_init<<<dim3(1), dim3(256), 0, stream>>>(selnum);
    k_masks<<<dim3(N_), dim3(256), 0, stream>>>(msk, seg, m_text, m_ev, m_claim, claimsum);
    k_norm<<<dim3(N_ * L_), dim3(256), 0, stream>>>(hiddens, hn);
    k_pairpool<<<dim3(QT, EVI * EVI, B_), dim3(32), 0, stream>>>(
        hn, m_text, m_ev, m_claim, W_att, b_att, W_sel, score, selnum);
    k_attn_denoise<<<dim3(B_ * EVI * EVI), dim3(256), 0, stream>>>(score, hiddens, denoise);
    k_gate<<<dim3(B_ * EVI * EVI), dim3(128), 0, stream>>>(pooled, denoise, Wg1, bg1, Wg2, bg2, gate);
    k_selfinal<<<dim3(1), dim3(256), 0, stream>>>(selnum, claimsum, b_sel, selprob);
    k_decombine<<<dim3(N_), dim3(256), 0, stream>>>(gate, denoise, de);
    k_feat<<<dim3(N_), dim3(96), 0, stream>>>(pooled, de, Wd, bd, feat);
    k_final<<<dim3(B_), dim3(32), 0, stream>>>(feat, selprob, out);
}
